// MultiheadAttention_78168404787682
// MI455X (gfx1250) — matrix operand order repaired and checked by reading, not compiled, not run on hardware
//
#include <hip/hip_runtime.h>

// ---------------------------------------------------------------------------
// MultiheadAttention forward for MI455X (gfx1250), bf16 WMMA everywhere,
// async global->LDS staging (ASYNCcnt) + TDM tensor tile load (TENSORcnt).
// B=4, TQ=TK=2048, D=1024, H=16, hd=64.
// ---------------------------------------------------------------------------

typedef unsigned short ushort_t;
typedef __attribute__((ext_vector_type(16))) __bf16       v16bf;
typedef __attribute__((ext_vector_type(8)))  float        v8f;
typedef __attribute__((ext_vector_type(8)))  ushort_t     v8us;
typedef __attribute__((ext_vector_type(4)))  int          v4i;
typedef __attribute__((ext_vector_type(4)))  unsigned int v4u;
typedef __attribute__((ext_vector_type(8)))  int          v8i;

#define BATCH   4
#define TSEQ    2048
#define DMODEL  1024
#define NHEADS  16
#define HDIM    64

// ---- async global->LDS (CDNA5 GLOBAL_LOAD_ASYNC_TO_LDS_B128, ASYNCcnt) ----
// Signature (from clang diagnostic): (v4i addrspace(1)*, v4i addrspace(3)*, Imm, Imm)
#if __has_builtin(__builtin_amdgcn_global_load_async_to_lds_b128)
#define HAVE_ASYNC 1
#define ASYNC_B128(ldsp, gp)                                                   \
  __builtin_amdgcn_global_load_async_to_lds_b128(                              \
      (__attribute__((address_space(1))) v4i*)(gp),                            \
      (__attribute__((address_space(3))) v4i*)(ldsp), 0, 0)
#else
#define HAVE_ASYNC 0
#endif

__device__ __forceinline__ void wait_async0() {
#if __has_builtin(__builtin_amdgcn_s_wait_asynccnt)
  __builtin_amdgcn_s_wait_asynccnt(0);
#else
  asm volatile("s_wait_asynccnt 0x0" ::: "memory");
#endif
}

__device__ __forceinline__ void wait_tensor0() {
#if __has_builtin(__builtin_amdgcn_s_wait_tensorcnt)
  __builtin_amdgcn_s_wait_tensorcnt(0);
#else
  asm volatile("s_wait_tensorcnt 0x0" ::: "memory");
#endif
}

// ---- TDM: 2D bf16 tile (rows x 64) global -> LDS via Tensor Data Mover ----
// D# per CDNA5 ISA ch.8: group0 = {flags, lds_addr, gaddr_lo, gaddr_hi|type},
// group1 = {mask/dsize, dims/strides/tile dims}, groups 2-3 zero for 2D.
// clang-23 builtin arity is 6: (v4u, v8i, v4i, v4i, v8i, Imm cpol).
#if __has_builtin(__builtin_amdgcn_tensor_load_to_lds)
#define HAVE_TDM 1
__device__ __forceinline__ void tdm_load_tile_2d(const ushort_t* gsrc,
                                                 ushort_t* ldst,
                                                 int rows_total) {
  unsigned lds_addr =
      (unsigned)(size_t)(__attribute__((address_space(3))) ushort_t*)ldst;
  unsigned long long ga = (unsigned long long)(size_t)gsrc;
  v4u g0 = { 1u,                                            // count=1 (valid)
             lds_addr,                                      // [63:32] lds_addr
             (unsigned)(ga & 0xFFFFFFFFu),                  // [95:64] gaddr lo
             (unsigned)((ga >> 32) & 0x1FFFFFFu) | (2u << 30) };  // type=2
  v8i g1 = { (int)0x00010000,          // wg_mask=0, data_size=1 (2 bytes)
             (int)(HDIM << 16),        // [63:48] tensor_dim0 = 64
             (int)(rows_total << 16),  // [95:80] tensor_dim1 lo16
             (int)(HDIM << 16),        // [127:112] tile_dim0 = 64
             64,                       // [143:128] tile_dim1 = 64, tile_dim2=0
             HDIM,                     // [191:160] tensor_dim0_stride = 64
             0, 0 };
  v4i g2 = {0, 0, 0, 0};
  v4i g3 = {0, 0, 0, 0};
  v8i g4 = {0, 0, 0, 0, 0, 0, 0, 0};
  __builtin_amdgcn_tensor_load_to_lds(g0, g1, g2, g3, g4, 0);
}
#else
#define HAVE_TDM 0
#endif

__device__ __forceinline__ ushort_t f32_to_bf16(float f) {
  union { float f; unsigned u; } x; x.f = f;
  unsigned u = x.u;
  unsigned r = u + 0x7FFFu + ((u >> 16) & 1u);   // round-to-nearest-even
  return (ushort_t)(r >> 16);
}

// A-operand fragment (16x32 bf16, MxK). Lane L: row m = L%16;
// K = kh*16..+15 contiguous, kh = L/16.
__device__ __forceinline__ v16bf load_fragA(const ushort_t* base, int lane, int stride) {
  const int row = lane & 15;
  const int kh  = (lane >> 4) & 1;
  const ushort_t* p = base + row * stride + kh * 16;
  union { v8us u[2]; v16bf b; } cv;
  cv.u[0] = *(const v8us*)(p);
  cv.u[1] = *(const v8us*)(p + 8);
  return cv.b;
}

// B-operand fragment (32x16 bf16, KxN) from an N-major source (row n holds
// contiguous K). Lane L: n = L%16, K = kh*16..+15 contiguous.
__device__ __forceinline__ v16bf load_fragB(const ushort_t* base, int lane, int stride) {
  const int row = lane & 15;
  const int kh  = (lane >> 4) & 1;
  const ushort_t* p = base + row * stride + kh * 16;
  union { v8us u[2]; v16bf b; } cv;
  cv.u[0] = *(const v8us*)(p);
  cv.u[1] = *(const v8us*)(p + 8);
  return cv.b;
}

// B-operand fragment from a K-major source (row k holds contiguous N):
// element j = base[(kh*16 + j)*stride + n].
__device__ __forceinline__ v16bf load_fragB_strided(const ushort_t* base, int lane, int stride) {
  const int col = lane & 15;
  const int kh  = (lane >> 4) & 1;
  union { ushort_t s[16]; v16bf b; } cv;
#pragma unroll
  for (int j = 0; j < 16; ++j)
    cv.s[j] = base[(kh * 16 + j) * stride + col];
  return cv.b;
}

#define WMMA_BF16(a, b, c) \
  __builtin_amdgcn_wmma_f32_16x16x32_bf16(false, (a), false, (b), (short)0, (c), false, false)

// ---------------------------------------------------------------------------
// fp32 -> bf16 conversion / key-padding-mask folding
// ---------------------------------------------------------------------------
__global__ void cvt_f32_bf16(const float* __restrict__ src,
                             ushort_t* __restrict__ dst, int n) {
  int i = blockIdx.x * blockDim.x + threadIdx.x;
  int stride = gridDim.x * blockDim.x;
  for (; i < n; i += stride) dst[i] = f32_to_bf16(src[i]);
}

__global__ void make_padf(const unsigned char* __restrict__ kpm,
                          float* __restrict__ padf, int n) {
  int i = blockIdx.x * blockDim.x + threadIdx.x;
  if (i < n) padf[i] = kpm[i] ? -1e30f : 0.0f;
}

// ---------------------------------------------------------------------------
// GEMM: Y[m,n] = sum_k X[m,k] * W[n,k]  (X: MxK bf16, W: NxK bf16)
// Block tile 128x64, 8 waves (4x2), wave tile 32x32, BK=32, double-buffered
// async LDS staging.
// MODE 0: f32 out[m*N+n];  MODE 1: bf16 per-head out[((b*16+h)*T+t)*64+hd].
// ---------------------------------------------------------------------------
#define GB_M 128
#define GB_N 64
#define GB_K 32

template <int MODE>
__global__ __launch_bounds__(256) void gemm_bf16(
    const ushort_t* __restrict__ X, const ushort_t* __restrict__ W,
    float* __restrict__ outF, ushort_t* __restrict__ outH,
    int M, int N, int K, int Trows) {
  __shared__ ushort_t lX[2][GB_M * GB_K];
  __shared__ ushort_t lW[2][GB_N * GB_K];

  const int tid  = threadIdx.x;
  const int lane = tid & 31;
  const int wave = tid >> 5;
  const int wM   = wave >> 1;          // 0..3
  const int wN   = wave & 1;           // 0..1
  const int bM   = blockIdx.x * GB_M;
  const int bN   = blockIdx.y * GB_N;

  v8f acc[2][2] = {};

  const int xr = tid >> 1, xc = (tid & 1) * 16;   // X: 128 rows x 32
  const int wr = tid >> 2, wc = (tid & 3) * 8;    // W: 64 rows x 32

  auto stage = [&](int buf, int k0) {
    const ushort_t* xp = X + (size_t)(bM + xr) * K + k0 + xc;
    const ushort_t* wp = W + (size_t)(bN + wr) * K + k0 + wc;
#if HAVE_ASYNC
    ASYNC_B128(&lX[buf][xr * GB_K + xc],     xp);
    ASYNC_B128(&lX[buf][xr * GB_K + xc + 8], xp + 8);
    ASYNC_B128(&lW[buf][wr * GB_K + wc],     wp);
#else
    *(v8us*)(&lX[buf][xr * GB_K + xc])     = *(const v8us*)(xp);
    *(v8us*)(&lX[buf][xr * GB_K + xc + 8]) = *(const v8us*)(xp + 8);
    *(v8us*)(&lW[buf][wr * GB_K + wc])     = *(const v8us*)(wp);
#endif
  };

  stage(0, 0);
  wait_async0();
  __syncthreads();

  int buf = 0;
  for (int k0 = 0; k0 < K; k0 += GB_K) {
    if (k0 + GB_K < K) stage(buf ^ 1, k0 + GB_K);            // prefetch next
    __builtin_prefetch(X + (size_t)(bM + xr) * K + k0 + 2 * GB_K + xc, 0, 1);

    v16bf a0 = load_fragA(&lX[buf][(wM * 32 + 0)  * GB_K], lane, GB_K);
    v16bf a1 = load_fragA(&lX[buf][(wM * 32 + 16) * GB_K], lane, GB_K);
    v16bf b0 = load_fragB(&lW[buf][(wN * 32 + 0)  * GB_K], lane, GB_K);
    v16bf b1 = load_fragB(&lW[buf][(wN * 32 + 16) * GB_K], lane, GB_K);
    acc[0][0] = WMMA_BF16(a0, b0, acc[0][0]);
    acc[0][1] = WMMA_BF16(a0, b1, acc[0][1]);
    acc[1][0] = WMMA_BF16(a1, b0, acc[1][0]);
    acc[1][1] = WMMA_BF16(a1, b1, acc[1][1]);

    wait_async0();       // next buffer fully landed in LDS
    __syncthreads();     // everyone done reading cur + writing next
    buf ^= 1;
  }

  // C/D layout: VGPR r -> M = r + 8*(lane/16), N = lane%16
  const int mh = (lane >> 4) * 8;
  const int nl = lane & 15;
#pragma unroll
  for (int mt = 0; mt < 2; ++mt)
#pragma unroll
    for (int nt = 0; nt < 2; ++nt)
#pragma unroll
      for (int r = 0; r < 8; ++r) {
        int m = bM + wM * 32 + mt * 16 + mh + r;
        int n = bN + wN * 32 + nt * 16 + nl;
        float v = acc[mt][nt][r];
        if (MODE == 0) {
          outF[(size_t)m * N + n] = v;
        } else {
          int b = m / Trows, t = m % Trows;
          int h = n >> 6, hd = n & 63;
          outH[(((size_t)(b * NHEADS + h) * Trows) + t) * HDIM + hd] = f32_to_bf16(v);
        }
      }
}

// ---------------------------------------------------------------------------
// Flash attention: per block 128 q-rows of one (b,h); per wave 16 q-rows.
// K tile async-staged (ASYNCcnt), V tile DMA'd by the TDM (TENSORcnt);
// online softmax; P routed through per-wave LDS for the A-frag reload.
// ---------------------------------------------------------------------------
__global__ __launch_bounds__(256) void flash_attn(
    const ushort_t* __restrict__ Qh, const ushort_t* __restrict__ Kh,
    const ushort_t* __restrict__ Vh, const float* __restrict__ amask,
    const float* __restrict__ padf, ushort_t* __restrict__ Ao) {
  __shared__ ushort_t lK[64 * 64];           // [k][d] row-major
  __shared__ ushort_t lV[64 * 64];           // [k][d] row-major
  __shared__ ushort_t lP[8][16 * 64];        // per-wave P scratch

  const int tid  = threadIdx.x;
  const int lane = tid & 31;
  const int wave = tid >> 5;
  const int bh   = blockIdx.y;               // b*16 + h
  const int b    = bh >> 4;
  const int q0   = blockIdx.x * 128 + wave * 16;
  const size_t headOff = (size_t)bh * TSEQ * HDIM;

  const int qrow = lane & 15;
  const int kh   = lane >> 4;

  // Q A-fragments held in registers; d = 64 -> 2 frags
  v16bf qa[2];
  {
    const ushort_t* qp = Qh + headOff + (size_t)(q0 + qrow) * HDIM;
#pragma unroll
    for (int f = 0; f < 2; ++f) {
      union { v8us u[2]; v16bf b; } cv;
      cv.u[0] = *(const v8us*)(qp + f * 32 + kh * 16);
      cv.u[1] = *(const v8us*)(qp + f * 32 + kh * 16 + 8);
      qa[f] = cv.b;
    }
  }

  v8f oacc[4] = {};                           // 4 d-subtiles of 16
  float mrun[8], lrun[8];
#pragma unroll
  for (int r = 0; r < 8; ++r) { mrun[r] = -1e30f; lrun[r] = 0.f; }

  const int sr = tid >> 2;                    // stager: k row (0..63)
  const int sc = (tid & 3) * 16;              // 16 d values
  const float log2e  = 1.44269504088896340736f;
  const float sscale = 0.125f;                // 1/sqrt(64)

  for (int kb = 0; kb < TSEQ; kb += 64) {
    // ---- stage K (async per-lane) and V (TDM tile DMA) into LDS ----
    {
      const ushort_t* kp = Kh + headOff + (size_t)(kb + sr) * HDIM + sc;
#if HAVE_ASYNC
      ASYNC_B128(&lK[sr * 64 + sc],     kp);
      ASYNC_B128(&lK[sr * 64 + sc + 8], kp + 8);
#else
      *(v8us*)(&lK[sr * 64 + sc])     = *(const v8us*)(kp);
      *(v8us*)(&lK[sr * 64 + sc + 8]) = *(const v8us*)(kp + 8);
#endif
      const ushort_t* vp = Vh + headOff + (size_t)(kb + sr) * HDIM + sc;
#if HAVE_TDM
      (void)vp;
      if (wave == 0)    // one TDM descriptor moves the whole 64x64 tile
        tdm_load_tile_2d(Vh + headOff + (size_t)kb * HDIM, &lV[0], TSEQ - kb);
#elif HAVE_ASYNC
      ASYNC_B128(&lV[sr * 64 + sc],     vp);
      ASYNC_B128(&lV[sr * 64 + sc + 8], vp + 8);
#else
      *(v8us*)(&lV[sr * 64 + sc])     = *(const v8us*)(vp);
      *(v8us*)(&lV[sr * 64 + sc + 8]) = *(const v8us*)(vp + 8);
#endif
    }
    wait_async0();
    wait_tensor0();      // free for waves with TENSORcnt==0
    __syncthreads();

    // ---- S = Q K^T : 4 k-subtiles x 2 d-halves = 8 WMMAs ----
    v8f sacc[4];
#pragma unroll
    for (int nt = 0; nt < 4; ++nt) {
      v8f s = {};
#pragma unroll
      for (int f = 0; f < 2; ++f) {
        v16bf kbf = load_fragB(&lK[(nt * 16) * 64] + f * 32, lane, 64);
        s = WMMA_BF16(qa[f], kbf, s);
      }
      sacc[nt] = s;
    }

    // ---- scale + masks + online softmax ----
#pragma unroll
    for (int r = 0; r < 8; ++r) {
      int q = q0 + kh * 8 + r;
      float mx = -1e30f;
#pragma unroll
      for (int nt = 0; nt < 4; ++nt) {
        int k = kb + nt * 16 + (lane & 15);
        float v = sacc[nt][r] * sscale + amask[(size_t)q * TSEQ + k]
                + padf[(size_t)b * TSEQ + k];
        sacc[nt][r] = v;
        mx = fmaxf(mx, v);
      }
      for (int d = 1; d < 16; d <<= 1) mx = fmaxf(mx, __shfl_xor(mx, d, 32));
      float mnew = fmaxf(mrun[r], mx);
      float corr = exp2f((mrun[r] - mnew) * log2e);
      mrun[r] = mnew;
      lrun[r] *= corr;
      float rowsum = 0.f;
#pragma unroll
      for (int nt = 0; nt < 4; ++nt) {
        float p = exp2f((sacc[nt][r] - mnew) * log2e);
        sacc[nt][r] = p;
        rowsum += p;
      }
      for (int d = 1; d < 16; d <<= 1) rowsum += __shfl_xor(rowsum, d, 32);
      lrun[r] += rowsum;
#pragma unroll
      for (int dt = 0; dt < 4; ++dt) oacc[dt][r] *= corr;
    }

    // ---- P (C-layout) -> per-wave LDS row-major 16x64 for A-frag reload ----
    ushort_t* pw = &lP[wave][0];
#pragma unroll
    for (int nt = 0; nt < 4; ++nt)
#pragma unroll
      for (int r = 0; r < 8; ++r)
        pw[(kh * 8 + r) * 64 + nt * 16 + (lane & 15)] = f32_to_bf16(sacc[nt][r]);

    v16bf pa[2];
#pragma unroll
    for (int f = 0; f < 2; ++f) pa[f] = load_fragA(pw + f * 32, lane, 64);

    // ---- O += P V : 4 d-subtiles x 2 k-halves = 8 WMMAs ----
#pragma unroll
    for (int dt = 0; dt < 4; ++dt)
#pragma unroll
      for (int f = 0; f < 2; ++f) {
        v16bf vb = load_fragB_strided(&lV[(f * 32) * 64 + dt * 16], lane, 64);
        oacc[dt] = WMMA_BF16(pa[f], vb, oacc[dt]);
      }
    __syncthreads();
  }

  // ---- normalize + store to (b, t, h*64+d) bf16 ----
  const int h = bh & 15;
#pragma unroll
  for (int r = 0; r < 8; ++r) {
    int q = q0 + kh * 8 + r;
    float inv = 1.0f / lrun[r];
#pragma unroll
    for (int dt = 0; dt < 4; ++dt) {
      int d = dt * 16 + (lane & 15);
      Ao[((size_t)b * TSEQ + q) * DMODEL + h * HDIM + d] =
          f32_to_bf16(oacc[dt][r] * inv);
    }
  }
}

// ---------------------------------------------------------------------------
// Launch: cvt/padf -> 3 projections -> flash attention -> output projection
// ---------------------------------------------------------------------------
extern "C" void kernel_launch(void* const* d_in, const int* in_sizes, int n_in,
                              void* d_out, int out_size, void* d_ws, size_t ws_size,
                              hipStream_t stream) {
  (void)in_sizes; (void)n_in; (void)out_size; (void)ws_size;
  const float* x_q  = (const float*)d_in[0];
  const float* x_kv = (const float*)d_in[1];
  const float* amask = (const float*)d_in[2];
  const unsigned char* kpm = (const unsigned char*)d_in[3];  // jnp bool = 1 byte
  const float* Wq = (const float*)d_in[4];
  const float* Wk = (const float*)d_in[5];
  const float* Wv = (const float*)d_in[6];
  const float* Wo = (const float*)d_in[7];

  const size_t XE = (size_t)BATCH * TSEQ * DMODEL;  // 8,388,608
  const size_t WE = (size_t)DMODEL * DMODEL;        // 1,048,576

  ushort_t* ws    = (ushort_t*)d_ws;                // ~104 MB + 32 KB
  ushort_t* xq_b  = ws;
  ushort_t* xkv_b = xq_b  + XE;
  ushort_t* wq_b  = xkv_b + XE;
  ushort_t* wk_b  = wq_b  + WE;
  ushort_t* wv_b  = wk_b  + WE;
  ushort_t* wo_b  = wv_b  + WE;
  ushort_t* Qh    = wo_b  + WE;
  ushort_t* Kh    = Qh    + XE;
  ushort_t* Vh    = Kh    + XE;
  ushort_t* Ao    = Vh    + XE;
  float*    padf  = (float*)(Ao + XE);

  cvt_f32_bf16<<<2048, 256, 0, stream>>>(x_q,  xq_b,  (int)XE);
  cvt_f32_bf16<<<2048, 256, 0, stream>>>(x_kv, xkv_b, (int)XE);
  cvt_f32_bf16<<<1024, 256, 0, stream>>>(Wq, wq_b, (int)WE);
  cvt_f32_bf16<<<1024, 256, 0, stream>>>(Wk, wk_b, (int)WE);
  cvt_f32_bf16<<<1024, 256, 0, stream>>>(Wv, wv_b, (int)WE);
  cvt_f32_bf16<<<1024, 256, 0, stream>>>(Wo, wo_b, (int)WE);
  make_padf<<<(BATCH * TSEQ) / 256, 256, 0, stream>>>(kpm, padf, BATCH * TSEQ);

  const int M = BATCH * TSEQ;        // 8192
  dim3 gg(M / GB_M, DMODEL / GB_N);  // 64 x 16
  gemm_bf16<1><<<gg, 256, 0, stream>>>(xq_b,  wq_b, nullptr, Qh, M, DMODEL, DMODEL, TSEQ);
  gemm_bf16<1><<<gg, 256, 0, stream>>>(xkv_b, wk_b, nullptr, Kh, M, DMODEL, DMODEL, TSEQ);
  gemm_bf16<1><<<gg, 256, 0, stream>>>(xkv_b, wv_b, nullptr, Vh, M, DMODEL, DMODEL, TSEQ);

  flash_attn<<<dim3(TSEQ / 128, BATCH * NHEADS), 256, 0, stream>>>(
      Qh, Kh, Vh, amask, padf, Ao);

  gemm_bf16<0><<<gg, 256, 0, stream>>>(Ao, wo_b, (float*)d_out, nullptr, M, DMODEL, DMODEL, TSEQ);
}
